// SpBasicBlock_88373247082547
// MI455X (gfx1250) — matrix operand order repaired and checked by reading, not compiled, not run on hardware
//
#include <hip/hip_runtime.h>

#define NROWS  262144
#define CCH    64
#define KOFF   27
#define PPAIRS 131072
#define EPSV   1e-5f
#define TILES_PER_WAVE 32
#define WLAY_ELEMS (KOFF * 2 * 4 * 32 * 16)   // 110592 bf16 per W

typedef __attribute__((ext_vector_type(16))) __bf16 v16bf;
typedef __attribute__((ext_vector_type(8)))  __bf16 v8bf;
typedef __attribute__((ext_vector_type(4)))  __bf16 v4bf;
typedef __attribute__((ext_vector_type(8)))  float  v8f;
typedef __attribute__((ext_vector_type(4)))  float  v4f;
typedef __attribute__((ext_vector_type(4)))  int    v4i;

__device__ __forceinline__ void atomAddF(float* p, float v) {
  __hip_atomic_fetch_add(p, v, __ATOMIC_RELAXED, __HIP_MEMORY_SCOPE_AGENT);
}

// ---------------------------------------------------------------------------
// Repack W (f32 [27][64][64], row=in-ch, col=out-ch) into bf16 tiles in the
// per-lane B-matrix layout: tile (k, kc, nt), lane L, element e holds
// W[k][kc*32 + (L>>4)*16 + e][nt*16 + (L&15)].
// ---------------------------------------------------------------------------
__global__ __launch_bounds__(256) void prep_w_kernel(
    const float* __restrict__ W1, const float* __restrict__ W2,
    __bf16* __restrict__ w1lay, __bf16* __restrict__ w2lay) {
  int idx = blockIdx.x * blockDim.x + threadIdx.x;
  if (idx >= 2 * WLAY_ELEMS) return;
  const float* W = (idx < WLAY_ELEMS) ? W1 : W2;
  __bf16* out = (idx < WLAY_ELEMS) ? w1lay : w2lay;
  int i = (idx < WLAY_ELEMS) ? idx : idx - WLAY_ELEMS;
  int e    = i & 15;
  int lane = (i >> 4) & 31;
  int nt   = (i >> 9) & 3;
  int kc   = (i >> 11) & 1;
  int k    = i >> 12;
  int row = kc * 32 + ((lane >> 4) << 4) + e;   // input channel (K dim)
  int col = nt * 16 + (lane & 15);              // output channel (N dim)
  out[i] = (__bf16)W[(size_t)k * 4096 + (size_t)row * 64 + col];
}

// Convert x (f32, row-major [N][64]) to a bf16 copy once.
__global__ __launch_bounds__(256) void prep_x_kernel(
    const float* __restrict__ x, __bf16* __restrict__ xbf) {
  int t = blockIdx.x * blockDim.x + threadIdx.x;   // float4 index
  v4f v = ((const v4f*)x)[t];
  v4bf b;
#pragma unroll
  for (int j = 0; j < 4; ++j) b[j] = (__bf16)v[j];
  ((v4bf*)xbf)[t] = b;
}

__global__ __launch_bounds__(256) void zero_kernel(
    float* __restrict__ acc, float* __restrict__ stats) {
  int t = blockIdx.x * blockDim.x + threadIdx.x;
  v4f z = {0.f, 0.f, 0.f, 0.f};
  if (t < (NROWS * CCH) / 4) ((v4f*)acc)[t] = z;
  if (t < 64) ((v4f*)stats)[t] = z;              // stats1 (128) + stats2 (128)
}

// ---------------------------------------------------------------------------
// Sparse conv: bf16 gather -> WMMA -> atomic scatter-add (f32).
// grid = (27, 32), block = 256 (8 waves). Each wave: 32 tiles of 16 pairs.
// ---------------------------------------------------------------------------
__global__ __launch_bounds__(256) void spconv_kernel(
    const __bf16* __restrict__ xin, const __bf16* __restrict__ wlay,
    const int* __restrict__ in_idx, const int* __restrict__ out_idx,
    float* __restrict__ acc) {
  const int k    = blockIdx.x;
  const int wave = threadIdx.x >> 5;
  const int lane = threadIdx.x & 31;
  const int r    = lane & 15;      // M (A) / N (B,D) position
  const int hi   = lane >> 4;      // lane half
  const int chunk = blockIdx.y * 8 + wave;   // 0..255

  // Load the 8 B tiles (2 K-chunks x 4 N-tiles) once; reused for 512 pairs.
  v16bf B[2][4];
  const v16bf* wb = (const v16bf*)(wlay + (size_t)k * 4096);
#pragma unroll
  for (int kc = 0; kc < 2; ++kc)
#pragma unroll
    for (int nt = 0; nt < 4; ++nt)
      B[kc][nt] = wb[(kc * 4 + nt) * 32 + lane];

  const int* iidx = in_idx  + (size_t)k * PPAIRS;
  const int* oidx = out_idx + (size_t)k * PPAIRS;
  const int tbase = chunk * TILES_PER_WAVE;

  for (int t = 0; t < TILES_PER_WAVE; ++t) {
    const int pb = (tbase + t) * 16;

    // ---- gather A: lane (r,hi) loads row in_idx[pb+r], its K-subset ----
    unsigned gi = (unsigned)iidx[pb + r];
    const __bf16* src = xin + gi * 64u;          // 32-bit offset -> saddr form
    union { v16bf v; v8bf h[2]; } a0, a1;
    a0.h[0] = *(const v8bf*)(src + hi * 16);         // K = hi*16  .. hi*16+7
    a0.h[1] = *(const v8bf*)(src + 8 + hi * 16);     // K = hi*16+8..hi*16+15
    a1.h[0] = *(const v8bf*)(src + 32 + hi * 16);    // chunk1
    a1.h[1] = *(const v8bf*)(src + 40 + hi * 16);

    // ---- 8 WMMAs: D[16x64] = A[16x64] * Wk[64x64] ----
    v8f d[4];
#pragma unroll
    for (int nt = 0; nt < 4; ++nt) {
      v8f z = {};
      z = __builtin_amdgcn_wmma_f32_16x16x32_bf16(false, a0.v, false, B[0][nt],
                                                  (short)0, z, false, false);
      z = __builtin_amdgcn_wmma_f32_16x16x32_bf16(false, a1.v, false, B[1][nt],
                                                  (short)0, z, false, false);
      d[nt] = z;
    }

    // ---- scatter-add: D VGPR i -> row out_idx[pb + hi*8 + i], col nt*16+r ----
    v4i o0 = *(const v4i*)(oidx + pb + hi * 8);
    v4i o1 = *(const v4i*)(oidx + pb + hi * 8 + 4);
    int rows[8] = {o0.x, o0.y, o0.z, o0.w, o1.x, o1.y, o1.z, o1.w};
#pragma unroll
    for (int i = 0; i < 8; ++i) {
      unsigned ro = (unsigned)rows[i] * 64u + (unsigned)r;  // u32 -> saddr form
      float* dst = acc + ro;
#pragma unroll
      for (int nt = 0; nt < 4; ++nt)
        atomAddF(dst + nt * 16, d[nt][i]);
    }
  }
}

// ---------------------------------------------------------------------------
// Per-channel sum / sum-of-squares. grid=256 blocks, each covers 1024 rows.
// ---------------------------------------------------------------------------
__global__ __launch_bounds__(256) void stats_kernel(
    const float* __restrict__ acc, float* __restrict__ stats) {
  int c  = threadIdx.x & 63;
  int rg = threadIdx.x >> 6;                 // 0..3
  size_t rowBase = (size_t)blockIdx.x * 1024;
  float s = 0.f, q = 0.f;
  for (int rr = rg; rr < 1024; rr += 4) {
    float v = acc[(rowBase + rr) * CCH + c];
    s += v; q += v * v;
  }
  __shared__ float sm[4][64];
  __shared__ float qm[4][64];
  sm[rg][c] = s; qm[rg][c] = q;
  __syncthreads();
  if (rg == 0) {
    s = sm[0][c] + sm[1][c] + sm[2][c] + sm[3][c];
    q = qm[0][c] + qm[1][c] + qm[2][c] + qm[3][c];
    atomAddF(&stats[c], s);
    atomAddF(&stats[64 + c], q);
  }
}

// BN + ReLU -> hbuf (bf16, ready for the next conv's A gather);
// also zero acc for reuse as the second accumulator.
__global__ __launch_bounds__(256) void bn_relu_zero_kernel(
    float* __restrict__ acc, const float* __restrict__ stats,
    const float* __restrict__ gamma, const float* __restrict__ beta,
    __bf16* __restrict__ hbuf) {
  int t = blockIdx.x * blockDim.x + threadIdx.x;   // float4 index
  int c0 = (t * 4) & 63;
  v4f v = ((const v4f*)acc)[t];
  v4bf h;
#pragma unroll
  for (int j = 0; j < 4; ++j) {
    int c = c0 + j;
    float mean = stats[c] * (1.f / NROWS);
    float var  = stats[64 + c] * (1.f / NROWS) - mean * mean;
    float sc   = gamma[c] * rsqrtf(var + EPSV);
    h[j] = (__bf16)fmaxf((v[j] - mean) * sc + beta[c], 0.f);
  }
  ((v4bf*)hbuf)[t] = h;
  ((v4f*)acc)[t] = (v4f){0.f, 0.f, 0.f, 0.f};
}

// BN + shortcut + ReLU -> d_out (f32).
__global__ __launch_bounds__(256) void final_kernel(
    const float* __restrict__ acc, const float* __restrict__ stats,
    const float* __restrict__ gamma, const float* __restrict__ beta,
    const float* __restrict__ xin, float* __restrict__ out) {
  int t = blockIdx.x * blockDim.x + threadIdx.x;
  int c0 = (t * 4) & 63;
  v4f v  = ((const v4f*)acc)[t];
  v4f xv = ((const v4f*)xin)[t];
  v4f h;
#pragma unroll
  for (int j = 0; j < 4; ++j) {
    int c = c0 + j;
    float mean = stats[c] * (1.f / NROWS);
    float var  = stats[64 + c] * (1.f / NROWS) - mean * mean;
    float sc   = gamma[c] * rsqrtf(var + EPSV);
    h[j] = fmaxf((v[j] - mean) * sc + beta[c] + xv[j], 0.f);
  }
  ((v4f*)out)[t] = h;
}

extern "C" void kernel_launch(void* const* d_in, const int* in_sizes, int n_in,
                              void* d_out, int out_size, void* d_ws, size_t ws_size,
                              hipStream_t stream) {
  const float* x      = (const float*)d_in[0];
  const float* W1     = (const float*)d_in[1];
  // d_in[2] = b1: cancels inside batchnorm (mean shifts by b) -> unused
  const float* gamma1 = (const float*)d_in[3];
  const float* beta1  = (const float*)d_in[4];
  const float* W2     = (const float*)d_in[5];
  // d_in[6] = b2: unused (same reason)
  const float* gamma2 = (const float*)d_in[7];
  const float* beta2  = (const float*)d_in[8];
  const int*   in_idx  = (const int*)d_in[9];
  const int*   out_idx = (const int*)d_in[10];
  float* out = (float*)d_out;

  // ws layout: acc f32[N*C] | stats f32[256] | w1lay,w2lay bf16 | xbf bf16[N*C] | hbuf bf16[N*C]
  float* acc   = (float*)d_ws;
  float* stats = acc + (size_t)NROWS * CCH;
  __bf16* w1lay = (__bf16*)(stats + 256);
  __bf16* w2lay = w1lay + WLAY_ELEMS;
  __bf16* xbf   = w2lay + WLAY_ELEMS;
  __bf16* hbuf  = xbf + (size_t)NROWS * CCH;

  const int elem4Blocks = (NROWS * CCH / 4) / 256;     // 16384

  prep_w_kernel<<<(2 * WLAY_ELEMS) / 256, 256, 0, stream>>>(W1, W2, w1lay, w2lay);
  prep_x_kernel<<<elem4Blocks, 256, 0, stream>>>(x, xbf);
  zero_kernel<<<elem4Blocks, 256, 0, stream>>>(acc, stats);

  spconv_kernel<<<dim3(KOFF, 32), 256, 0, stream>>>(xbf, w1lay, in_idx, out_idx, acc);
  stats_kernel<<<256, 256, 0, stream>>>(acc, stats);
  bn_relu_zero_kernel<<<elem4Blocks, 256, 0, stream>>>(acc, stats, gamma1, beta1, hbuf);

  spconv_kernel<<<dim3(KOFF, 32), 256, 0, stream>>>(hbuf, w2lay, in_idx, out_idx, acc);
  stats_kernel<<<256, 256, 0, stream>>>(acc, stats + 128);
  final_kernel<<<elem4Blocks, 256, 0, stream>>>(acc, stats + 128, gamma2, beta2, x, out);
}
